// WindowAttention_66322884985427
// MI455X (gfx1250) — matrix operand order repaired and checked by reading, not compiled, not run on hardware
//
#include <hip/hip_runtime.h>

typedef __attribute__((ext_vector_type(16))) _Float16 v16h;
typedef __attribute__((ext_vector_type(8)))  float    v8f;

#define WIN   64
#define REC   128
#define NH    8
#define HD    32
#define DIM   256
#define LDX   264   // padded row stride (halves) for 256-wide f16 buffers
#define LDM   136   // padded row stride for 128-wide buffers
#define QKSCALE 0.17677669529663687f  // 1/sqrt(32)

// ---------------- WMMA fragment loaders (wave32, 16x16x32 f16) ----------------
// A (16x32, MxK), row-major source with leading dim `ld` (halves):
//   lane<16 : elems0-7 = A[row0+lane][k0..k0+7],  elems8-15 = A[..][k0+16..k0+23]
//   lane>=16: elems0-7 = A[row0+l-16][k0+8..k0+15], elems8-15 = [k0+24..k0+31]
__device__ __forceinline__ v16h load_a(const _Float16* base, int ld, int row0, int k0) {
  const int lane = threadIdx.x & 31;
  const _Float16* p = base + (size_t)(row0 + (lane & 15)) * ld + k0 + ((lane >> 4) << 3);
  union { v16h h; uint4 q[2]; } u;
  u.q[0] = *(const uint4*)(p);
  u.q[1] = *(const uint4*)(p + 16);
  return u.h;
}

// B (32x16, KxN) loaded from transposed image Bt[N][K] with leading dim `ld`:
//   lane<16 : elems0-7 = B[k0..k0+7][col0+lane], elems8-15 = B[k0+16..k0+23] ; lane>=16: k0+8..k0+15, k0+24..k0+31
__device__ __forceinline__ v16h load_b(const _Float16* bt, int ld, int k0, int col0) {
  const int lane = threadIdx.x & 31;
  const _Float16* p = bt + (size_t)(col0 + (lane & 15)) * ld + k0 + ((lane >> 4) << 3);
  union { v16h h; uint4 q[2]; } u;
  u.q[0] = *(const uint4*)(p);
  u.q[1] = *(const uint4*)(p + 16);
  return u.h;
}

#define WMMA_F16(A, B, C) \
  __builtin_amdgcn_wmma_f32_16x16x32_f16(false, (A), false, (B), (short)0, (C), false, false)

// ---------------- prep: transpose + f16-convert weights into ws ----------------
// ws layout (halves): [0)      qwT  256x256  (out-major, in contiguous)
//                     [65536)  kvwT 512x256
//                     [196608) pwT  256x256
__global__ void wa_prep(const float* __restrict__ qw, const float* __restrict__ kvw,
                        const float* __restrict__ pw, _Float16* __restrict__ wsw) {
  int idx = blockIdx.x * blockDim.x + threadIdx.x;
  if (idx < 65536) {
    int o = idx >> 8, i = idx & 255;
    wsw[idx] = (_Float16)qw[i * 256 + o];
  } else if (idx < 65536 + 131072) {
    int j = idx - 65536; int o = j >> 8, i = j & 255;
    wsw[idx] = (_Float16)kvw[i * 512 + o];
  } else if (idx < 262144) {
    int j = idx - 196608; int o = j >> 8, i = j & 255;
    wsw[idx] = (_Float16)pw[i * 256 + o];
  }
}

// ---------------- fused per-window attention ----------------
struct SPArea {
  float    S[WIN * LDM];     // attention logits (f32)
  _Float16 P[WIN * LDM];     // softmax probs (f16)
  float    part[WIN * 8];    // [row][0..3]=max partials, [4..7]=sum partials
};
union WorkArea {
  _Float16 xbuf[REC * LDX];  // staging for x (64 rows) then x_ (128 rows)
  SPArea   sp;
};

__global__ __launch_bounds__(256, 1) void wa_main(
    const float* __restrict__ x, const float* __restrict__ x_,
    const int* __restrict__ mask_left, const int* __restrict__ mask_right,
    const int* __restrict__ nWp,
    const float* __restrict__ q_b, const float* __restrict__ kv_b,
    const float* __restrict__ proj_b, const float* __restrict__ bias_table,
    const _Float16* __restrict__ qwT, const _Float16* __restrict__ kvwT,
    const _Float16* __restrict__ pwT,
    float* __restrict__ out)
{
  __shared__ __align__(16) _Float16 Qbuf[WIN * LDX];        //  33.8 KB  [n][h*32+d]
  __shared__ __align__(16) _Float16 Kbuf[REC * LDX];        //  67.6 KB  [m][h*32+d]
  __shared__ __align__(16) _Float16 Vt[NH * HD * LDM];      //  69.6 KB  [h][d][m]
  __shared__ __align__(16) _Float16 Obuf[WIN * LDX];        //  33.8 KB  [n][h*32+d]
  __shared__ __align__(16) float    biasLds[192 * NH];      //   6.1 KB
  __shared__ __align__(16) float    bvec[1024];             //   4.0 KB  qb|kvb|pb
  __shared__ __align__(16) WorkArea un;                     //  67.6 KB

  const int tid  = threadIdx.x;
  const int wave = tid >> 5;
  const int lane = tid & 31;
  const int b    = blockIdx.x;

  // ---- stage constants ----
  for (int i = tid; i < 192 * NH; i += 256) biasLds[i] = bias_table[i];
  bvec[tid]       = q_b[tid];
  bvec[256 + tid] = kv_b[tid];
  bvec[512 + tid] = kv_b[256 + tid];
  bvec[768 + tid] = proj_b[tid];

  // ---- stage x (64x256) as f16 ----
  {
    const float4* xs = (const float4*)(x + (size_t)b * WIN * DIM);
    for (int i = tid; i < WIN * DIM / 4; i += 256) {
      float4 v = xs[i];
      int row = i >> 6, c4 = (i & 63) << 2;
      _Float16* d = un.xbuf + row * LDX + c4;
      d[0] = (_Float16)v.x; d[1] = (_Float16)v.y;
      d[2] = (_Float16)v.z; d[3] = (_Float16)v.w;
    }
  }
  __syncthreads();

  // ---- Q projection: 4x16 tiles of 16x16, K=256 ----
  #pragma unroll 1
  for (int t = wave; t < 64; t += 8) {
    const int ti = t >> 4, tj = t & 15;
    __builtin_prefetch((const void*)(qwT + (size_t)(tj * 16) * DIM), 0, 0);
    v8f acc = {};
    #pragma unroll
    for (int kb = 0; kb < 8; ++kb) {
      v16h a  = load_a(un.xbuf, LDX, ti * 16, kb * 32);
      v16h bf = load_b(qwT, DIM, kb * 32, tj * 16);
      acc = WMMA_F16(a, bf, acc);
    }
    const int col = tj * 16 + (lane & 15);
    const int n0  = ti * 16 + ((lane >> 4) << 3);
    const float bb = bvec[col];
    #pragma unroll
    for (int r = 0; r < 8; ++r) Qbuf[(n0 + r) * LDX + col] = (_Float16)(acc[r] + bb);
  }
  __syncthreads();

  // ---- stage x_ (128x256) as f16 (reuses xbuf) ----
  {
    const float4* xs = (const float4*)(x_ + (size_t)b * REC * DIM);
    for (int i = tid; i < REC * DIM / 4; i += 256) {
      float4 v = xs[i];
      int row = i >> 6, c4 = (i & 63) << 2;
      _Float16* d = un.xbuf + row * LDX + c4;
      d[0] = (_Float16)v.x; d[1] = (_Float16)v.y;
      d[2] = (_Float16)v.z; d[3] = (_Float16)v.w;
    }
  }
  __syncthreads();

  // ---- KV projection: 8x32 tiles; cols<256 -> K[m][c], cols>=256 -> Vt[h][d][m] ----
  #pragma unroll 1
  for (int t = wave; t < 256; t += 8) {
    const int ti = t >> 5, tj = t & 31;
    __builtin_prefetch((const void*)(kvwT + (size_t)(tj * 16) * DIM), 0, 0);
    v8f acc = {};
    #pragma unroll
    for (int kb = 0; kb < 8; ++kb) {
      v16h a  = load_a(un.xbuf, LDX, ti * 16, kb * 32);
      v16h bf = load_b(kvwT, DIM, kb * 32, tj * 16);
      acc = WMMA_F16(a, bf, acc);
    }
    const int ocol = tj * 16 + (lane & 15);
    const int m0   = ti * 16 + ((lane >> 4) << 3);
    const float bb = bvec[256 + ocol];
    if (ocol < 256) {
      #pragma unroll
      for (int r = 0; r < 8; ++r) Kbuf[(m0 + r) * LDX + ocol] = (_Float16)(acc[r] + bb);
    } else {
      _Float16* vt = Vt + (size_t)(ocol - 256) * LDM + m0;   // [h*32+d][m] contiguous in m
      #pragma unroll
      for (int r = 0; r < 8; ++r) vt[r] = (_Float16)(acc[r] + bb);
    }
  }
  __syncthreads();

  // ---- window mask selection ----
  const int nW = nWp[0];
  const int wi = b % nW;
  const int mc = nW < 2 ? nW : 2;
  const int* maskp = nullptr;
  if (wi >= nW - mc)      maskp = mask_right + (size_t)(wi - nW + 2) * WIN * REC;
  else if (wi < mc)       maskp = mask_left  + (size_t)wi * WIN * REC;

  // ---- per-head attention ----
  for (int h = 0; h < 8; ++h) {
    // S = Q_h (64x32) * K_h^T (32x128): 4x8 tiles, one WMMA each (K=32=head dim).
    {
      const int j = wave;                       // column block (key m)
      #pragma unroll 1
      for (int i = 0; i < 4; ++i) {             // row block (query n)
        v16h a  = load_a(Qbuf, LDX, i * 16, h * 32);
        v16h bf = load_b(Kbuf, LDX, h * 32, j * 16);
        v8f acc = {};
        acc = WMMA_F16(a, bf, acc);
        const int m  = j * 16 + (lane & 15);
        const int n0 = i * 16 + ((lane >> 4) << 3);
        #pragma unroll
        for (int r = 0; r < 8; ++r) {
          const int n = n0 + r;
          float s = acc[r] * QKSCALE + biasLds[(n - m + 128) * NH + h];
          if (maskp && maskp[n * REC + m] == 1) s = -3.0e38f;
          un.sp.S[n * LDM + m] = s;
        }
      }
    }
    __syncthreads();

    // Softmax over m=128: 4 threads per row, each owns 32 columns.
    {
      const int row = tid >> 2, qq = tid & 3;
      const float* srow = un.sp.S + row * LDM + qq * 32;
      float mx = -3.4e38f;
      #pragma unroll
      for (int c = 0; c < 32; ++c) mx = fmaxf(mx, srow[c]);
      un.sp.part[row * 8 + qq] = mx;
      __syncthreads();
      const float rmx = fmaxf(fmaxf(un.sp.part[row * 8 + 0], un.sp.part[row * 8 + 1]),
                              fmaxf(un.sp.part[row * 8 + 2], un.sp.part[row * 8 + 3]));
      float vals[32];
      float sum = 0.f;
      #pragma unroll
      for (int c = 0; c < 32; ++c) { float e = __expf(srow[c] - rmx); vals[c] = e; sum += e; }
      un.sp.part[row * 8 + 4 + qq] = sum;
      __syncthreads();
      const float inv = 1.0f / (un.sp.part[row * 8 + 4] + un.sp.part[row * 8 + 5] +
                                un.sp.part[row * 8 + 6] + un.sp.part[row * 8 + 7]);
      _Float16* prow = un.sp.P + row * LDM + qq * 32;
      #pragma unroll
      for (int c = 0; c < 32; ++c) prow[c] = (_Float16)(vals[c] * inv);
    }
    __syncthreads();

    // O_h = P (64x128) * V_h (128x32): 4x2 tiles, 1 per wave, K=128 (4 WMMA).
    {
      const int i = wave >> 1, jd = wave & 1;
      v8f acc = {};
      #pragma unroll
      for (int kb = 0; kb < 4; ++kb) {
        v16h a  = load_a(un.sp.P, LDM, i * 16, kb * 32);
        v16h bf = load_b(Vt + (size_t)h * HD * LDM, LDM, kb * 32, jd * 16);
        acc = WMMA_F16(a, bf, acc);
      }
      const int col = h * 32 + jd * 16 + (lane & 15);
      const int n0  = i * 16 + ((lane >> 4) << 3);
      #pragma unroll
      for (int r = 0; r < 8; ++r) Obuf[(n0 + r) * LDX + col] = (_Float16)acc[r];
    }
    __syncthreads();
  }

  // ---- output projection: out = O (64x256) * proj_w (256x256) + proj_b ----
  float* ob = out + (size_t)b * WIN * DIM;
  #pragma unroll 1
  for (int t = wave; t < 64; t += 8) {
    const int ti = t >> 4, tj = t & 15;
    __builtin_prefetch((const void*)(pwT + (size_t)(tj * 16) * DIM), 0, 0);
    v8f acc = {};
    #pragma unroll
    for (int kb = 0; kb < 8; ++kb) {
      v16h a  = load_a(Obuf, LDX, ti * 16, kb * 32);
      v16h bf = load_b(pwT, DIM, kb * 32, tj * 16);
      acc = WMMA_F16(a, bf, acc);
    }
    const int col = tj * 16 + (lane & 15);
    const int n0  = ti * 16 + ((lane >> 4) << 3);
    const float bb = bvec[768 + col];
    #pragma unroll
    for (int r = 0; r < 8; ++r) ob[(n0 + r) * DIM + col] = acc[r] + bb;
  }
}

extern "C" void kernel_launch(void* const* d_in, const int* in_sizes, int n_in,
                              void* d_out, int out_size, void* d_ws, size_t ws_size,
                              hipStream_t stream) {
  const float* x          = (const float*)d_in[0];
  const float* x_         = (const float*)d_in[1];
  const int*   mask_left  = (const int*)d_in[2];
  const int*   mask_right = (const int*)d_in[3];
  const int*   nWp        = (const int*)d_in[4];
  const float* q_w        = (const float*)d_in[5];
  const float* q_b        = (const float*)d_in[6];
  const float* kv_w       = (const float*)d_in[7];
  const float* kv_b       = (const float*)d_in[8];
  const float* proj_w     = (const float*)d_in[9];
  const float* proj_b     = (const float*)d_in[10];
  const float* bias_table = (const float*)d_in[11];

  _Float16* wsw = (_Float16*)d_ws;   // needs 512 KB: qwT | kvwT | pwT (f16)
  const int B_ = in_sizes[0] / (WIN * DIM);

  wa_prep<<<dim3(1024), dim3(256), 0, stream>>>(q_w, kv_w, proj_w, wsw);
  wa_main<<<dim3(B_), dim3(256), 0, stream>>>(
      x, x_, mask_left, mask_right, nWp, q_b, kv_b, proj_b, bias_table,
      wsw, wsw + 65536, wsw + 196608, (float*)d_out);
}
